// GATModel_9887014716235
// MI455X (gfx1250) — matrix operand order repaired and checked by reading, not compiled, not run on hardware
//
#include <hip/hip_runtime.h>
#include <hip/hip_bf16.h>

typedef __attribute__((ext_vector_type(16))) _Float16 v16h;
typedef __attribute__((ext_vector_type(8)))  _Float16 v8h;
typedef __attribute__((ext_vector_type(8)))  float    v8f;

#define IN_CH    256
#define HIDHEADS 256   // HEADS * HID
#define OUT_CH   64
#define NHEADS   4
#define NSLOPE   0.2f
#define SCAN_B   1024

// ---------------- conversion / transpose ----------------

__global__ void k_convert_x(const float* __restrict__ x, _Float16* __restrict__ xh, int n) {
  int i = blockIdx.x * blockDim.x + threadIdx.x;
  if (i < n) xh[i] = (_Float16)x[i];
}

// W is [K, Ncol] row-major; write Wt[o*K + k] = W[k*Ncol + o] as f16
__global__ void k_transpose_w(const float* __restrict__ W, _Float16* __restrict__ Wt,
                              int K, int Ncol) {
  int i = blockIdx.x * blockDim.x + threadIdx.x;
  if (i >= K * Ncol) return;
  int k = i / Ncol, o = i % Ncol;
  Wt[(size_t)o * K + k] = (_Float16)W[i];
}

// ---------------- CSR build (dst-sorted adjacency, shared by both layers) ----------------

__global__ void k_count(const long long* __restrict__ dstE, int E, int N,
                        int* __restrict__ deg) {
  int e = blockIdx.x * blockDim.x + threadIdx.x;
  if (e >= E + N) return;
  int d = (e < E) ? (int)dstE[e] : (e - E);
  atomicAdd(&deg[d], 1);
}

__global__ __launch_bounds__(SCAN_B) void k_scan1(const int* __restrict__ deg, int N,
                                                  int* __restrict__ incl,
                                                  int* __restrict__ bsum) {
  __shared__ int sb[SCAN_B];
  int tid = threadIdx.x;
  int i = blockIdx.x * SCAN_B + tid;
  sb[tid] = (i < N) ? deg[i] : 0;
  __syncthreads();
  for (int off = 1; off < SCAN_B; off <<= 1) {
    int t = (tid >= off) ? sb[tid - off] : 0;
    __syncthreads();
    sb[tid] += t;
    __syncthreads();
  }
  if (i < N) incl[i] = sb[tid];
  if (tid == SCAN_B - 1) bsum[blockIdx.x] = sb[SCAN_B - 1];
}

__global__ void k_scan2(const int* __restrict__ bsum, int* __restrict__ boff, int nb) {
  if (threadIdx.x == 0 && blockIdx.x == 0) {
    int run = 0;
    for (int j = 0; j < nb; ++j) { boff[j] = run; run += bsum[j]; }
  }
}

__global__ void k_scan3(const int* __restrict__ incl, const int* __restrict__ deg,
                        const int* __restrict__ boff, int N,
                        int* __restrict__ starts, int* __restrict__ cursor) {
  int i = blockIdx.x * blockDim.x + threadIdx.x;
  if (i >= N) return;
  int ex = incl[i] - deg[i] + boff[i >> 10];
  starts[i] = ex;
  cursor[i] = ex;
}

__global__ void k_scatter(const long long* __restrict__ srcE, const long long* __restrict__ dstE,
                          int E, int N, int* __restrict__ cursor,
                          int* __restrict__ csr_src, int* __restrict__ epos) {
  int e = blockIdx.x * blockDim.x + threadIdx.x;
  if (e >= E + N) return;
  int s, d;
  if (e < E) { s = (int)srcE[e]; d = (int)dstE[e]; } else { s = e - E; d = s; }
  int pos = atomicAdd(&cursor[d], 1);
  csr_src[pos] = s;
  epos[e] = pos;
}

// ---------------- WMMA GEMM: one wave computes a 16x64 tile (4 WMMAs share the A frag) -------
// A[M,K] f16 row-major, Bt[Ncol,K] f16 row-major (B transposed). K % 32 == 0, Ncol % 64 == 0.

__global__ __launch_bounds__(256) void k_gemm_wmma(
    const _Float16* __restrict__ A, const _Float16* __restrict__ Bt,
    const float* __restrict__ bias, float* __restrict__ C,
    int M, int K, int Ncol) {
  const int lane = threadIdx.x & 31;
  const int wave = threadIdx.x >> 5;
  const int ngrp = Ncol >> 6;                       // groups of 4 n-tiles
  const int t    = blockIdx.x * (blockDim.x >> 5) + wave;
  const int mt   = t / ngrp;
  const int ng   = t % ngrp;
  if (mt * 16 >= M) return;

  const int hi = lane >> 4;
  const int lo = lane & 15;
  const _Float16* arow = A + (size_t)(mt * 16 + lo) * K;
  const _Float16* brow[4];
#pragma unroll
  for (int j = 0; j < 4; ++j)
    brow[j] = Bt + (size_t)(ng * 64 + j * 16 + lo) * K;

  v8f acc[4] = {{}, {}, {}, {}};
  for (int kb = 0; kb < K; kb += 32) {
    v8h a0 = *(const v8h*)(arow + kb + hi * 16);
    v8h a1 = *(const v8h*)(arow + kb + hi * 16 + 8);
    v16h a = __builtin_shufflevector(a0, a1, 0,1,2,3,4,5,6,7,8,9,10,11,12,13,14,15);
#pragma unroll
    for (int j = 0; j < 4; ++j) {
      v16h b = *(const v16h*)(brow[j] + kb + hi * 16);
      acc[j] = __builtin_amdgcn_wmma_f32_16x16x32_f16(false, a, false, b, (short)0, acc[j],
                                                      false, false);
    }
  }
#pragma unroll
  for (int j = 0; j < 4; ++j) {
    const int n = ng * 64 + j * 16 + lo;
    const float bv = bias ? bias[n] : 0.0f;
    float* crow = C + (size_t)(mt * 16 + hi * 8) * Ncol + n;
#pragma unroll
    for (int v = 0; v < 8; ++v)
      crow[(size_t)v * Ncol] = acc[j][v] + bv;
  }
}

// ---------------- attention scores ----------------

__global__ void k_attn_scores(const float* __restrict__ h,
                              const float* __restrict__ att_src,
                              const float* __restrict__ att_dst,
                              float* __restrict__ a_src, float* __restrict__ a_dst,
                              int N, int heads, int ch) {
  int i = blockIdx.x * blockDim.x + threadIdx.x;
  if (i >= N * heads) return;
  int nd = i / heads, hd = i % heads;
  const float* hp = h + (size_t)nd * heads * ch + (size_t)hd * ch;
  const float* as = att_src + hd * ch;
  const float* ad = att_dst + hd * ch;
  float s0 = 0.f, s1 = 0.f;
  for (int c = 0; c < ch; ++c) { float v = hp[c]; s0 += v * as[c]; s1 += v * ad[c]; }
  a_src[i] = s0;
  a_dst[i] = s1;
}

// ---- edge exp into CSR order (no atomics); softmax max-shift is mathematically redundant ----

__global__ void k_edge_exp(const long long* __restrict__ srcE, const long long* __restrict__ dstE,
                           int E, int N, const float* __restrict__ a_src,
                           const float* __restrict__ a_dst, const int* __restrict__ epos,
                           float* __restrict__ csr_w, int heads) {
  int e = blockIdx.x * blockDim.x + threadIdx.x;
  if (e >= E + N) return;
  int s, d;
  if (e < E) { s = (int)srcE[e]; d = (int)dstE[e]; } else { s = e - E; d = s; }
  int pos = epos[e];
  for (int h = 0; h < heads; ++h) {
    float al = a_src[(size_t)s * heads + h] + a_dst[(size_t)d * heads + h];
    al = (al > 0.f) ? al : NSLOPE * al;
    csr_w[(size_t)pos * heads + h] = expf(al);
  }
}

// ---- layer-1 pull aggregation: one wave per node; fused bias + ReLU + f16 conversion --------

__global__ __launch_bounds__(256) void k_node_agg1(
    const int* __restrict__ starts, const int* __restrict__ deg,
    const int* __restrict__ csr_src, const float* __restrict__ csr_w,
    const float* __restrict__ h1, const float* __restrict__ b1,
    _Float16* __restrict__ h2in, int N) {
  int gw   = (int)((blockIdx.x * blockDim.x + threadIdx.x) >> 5);
  int lane = threadIdx.x & 31;
  if (gw >= N) return;
  const int st = starts[gw], dg = deg[gw];

  // per-head softmax denominator, lane-parallel over edges then wave-reduced
  float p0 = 0.f, p1 = 0.f, p2 = 0.f, p3 = 0.f;
  for (int j = st + lane; j < st + dg; j += 32) {
    const float* wp = csr_w + (size_t)j * NHEADS;
    p0 += wp[0]; p1 += wp[1]; p2 += wp[2]; p3 += wp[3];
  }
#pragma unroll
  for (int o = 16; o > 0; o >>= 1) {
    p0 += __shfl_xor(p0, o, 32); p1 += __shfl_xor(p1, o, 32);
    p2 += __shfl_xor(p2, o, 32); p3 += __shfl_xor(p3, o, 32);
  }
  const float inv[NHEADS] = { 1.f / (p0 + 1e-16f), 1.f / (p1 + 1e-16f),
                              1.f / (p2 + 1e-16f), 1.f / (p3 + 1e-16f) };

  float acc[8] = {0.f, 0.f, 0.f, 0.f, 0.f, 0.f, 0.f, 0.f};
  for (int j = st; j < st + dg; ++j) {
    const int s = csr_src[j];
    const float* wp = csr_w + (size_t)j * NHEADS;
    float cc[NHEADS];
#pragma unroll
    for (int h = 0; h < NHEADS; ++h) cc[h] = wp[h] * inv[h];
    const float* hs = h1 + (size_t)s * HIDHEADS;
#pragma unroll
    for (int k = 0; k < 8; ++k)
      acc[k] += hs[lane + 32 * k] * cc[k >> 1];   // head = channel/64 = k/2
  }
#pragma unroll
  for (int k = 0; k < 8; ++k) {
    const int ch = lane + 32 * k;
    float v = acc[k] + b1[ch];
    v = v > 0.f ? v : 0.f;                         // ReLU between layers
    h2in[(size_t)gw * HIDHEADS + ch] = (_Float16)v;
  }
}

// ---- layer-2 pull aggregation: fused bias + log_softmax, writes final output ---------------

__global__ __launch_bounds__(256) void k_node_agg2(
    const int* __restrict__ starts, const int* __restrict__ deg,
    const int* __restrict__ csr_src, const float* __restrict__ csr_w,
    const float* __restrict__ h2, const float* __restrict__ b2,
    float* __restrict__ out, int N) {
  int gw   = (int)((blockIdx.x * blockDim.x + threadIdx.x) >> 5);
  int lane = threadIdx.x & 31;
  if (gw >= N) return;
  const int st = starts[gw], dg = deg[gw];

  float p = 0.f;
  for (int j = st + lane; j < st + dg; j += 32) p += csr_w[j];
#pragma unroll
  for (int o = 16; o > 0; o >>= 1) p += __shfl_xor(p, o, 32);
  const float inv = 1.f / (p + 1e-16f);

  float a0 = 0.f, a1 = 0.f;
  for (int j = st; j < st + dg; ++j) {
    const int s = csr_src[j];
    const float c = csr_w[j] * inv;
    const float* hs = h2 + (size_t)s * OUT_CH;
    a0 += hs[lane] * c;
    a1 += hs[lane + 32] * c;
  }
  float v0 = a0 + b2[lane];
  float v1 = a1 + b2[lane + 32];
  float mx = fmaxf(v0, v1);
#pragma unroll
  for (int o = 16; o > 0; o >>= 1) mx = fmaxf(mx, __shfl_xor(mx, o, 32));
  float sum = expf(v0 - mx) + expf(v1 - mx);
#pragma unroll
  for (int o = 16; o > 0; o >>= 1) sum += __shfl_xor(sum, o, 32);
  const float lg = mx + logf(sum);
  out[(size_t)gw * OUT_CH + lane]      = v0 - lg;
  out[(size_t)gw * OUT_CH + lane + 32] = v1 - lg;
}

// ---------------- launch ----------------

extern "C" void kernel_launch(void* const* d_in, const int* in_sizes, int n_in,
                              void* d_out, int out_size, void* d_ws, size_t ws_size,
                              hipStream_t stream) {
  const float*     x   = (const float*)d_in[0];
  const long long* ei  = (const long long*)d_in[1];   // int64 [2, E]
  const float*     W1  = (const float*)d_in[2];
  const float*     as1 = (const float*)d_in[3];
  const float*     ad1 = (const float*)d_in[4];
  const float*     b1  = (const float*)d_in[5];
  const float*     W2  = (const float*)d_in[6];
  const float*     as2 = (const float*)d_in[7];
  const float*     ad2 = (const float*)d_in[8];
  const float*     b2  = (const float*)d_in[9];

  const int N    = in_sizes[0] / IN_CH;
  const int E    = in_sizes[1] / 2;
  const int Etot = E + N;
  const int nb   = (N + SCAN_B - 1) / SCAN_B;
  const long long* srcE = ei;
  const long long* dstE = ei + E;

  char* p = (char*)d_ws;
  auto carve = [&](size_t bytes) -> char* {
    char* r = p;
    p += (bytes + 255) & ~(size_t)255;
    return r;
  };
  _Float16* xh      = (_Float16*)carve((size_t)N * IN_CH * 2);        // reused as h2in
  float*    h1      = (float*)   carve((size_t)N * HIDHEADS * 4);
  _Float16* W1t     = (_Float16*)carve((size_t)IN_CH * HIDHEADS * 2);
  _Float16* W2t     = (_Float16*)carve((size_t)HIDHEADS * OUT_CH * 2);
  float*    a_src1  = (float*)   carve((size_t)N * NHEADS * 4);
  float*    a_dst1  = (float*)   carve((size_t)N * NHEADS * 4);
  float*    a_src2  = (float*)   carve((size_t)N * 4);
  float*    a_dst2  = (float*)   carve((size_t)N * 4);
  int*      deg     = (int*)     carve((size_t)N * 4);
  int*      incl    = (int*)     carve((size_t)N * 4);
  int*      starts  = (int*)     carve((size_t)N * 4);
  int*      cursor  = (int*)     carve((size_t)N * 4);
  int*      bsum    = (int*)     carve((size_t)nb * 4);
  int*      boff    = (int*)     carve((size_t)nb * 4);
  int*      epos    = (int*)     carve((size_t)Etot * 4);
  int*      csr_src = (int*)     carve((size_t)Etot * 4);
  float*    csr_w1  = (float*)   carve((size_t)Etot * NHEADS * 4);
  float*    csr_w2  = (float*)   carve((size_t)Etot * 4);
  float*    h2      = (float*)   carve((size_t)N * OUT_CH * 4);
  float*    outf    = (float*)d_out;

  hipMemsetAsync(deg, 0, (size_t)N * 4, stream);

  const int T = 256;

  // dense prep
  k_convert_x  <<<(N * IN_CH + T - 1) / T, T, 0, stream>>>(x, xh, N * IN_CH);
  k_transpose_w<<<(IN_CH * HIDHEADS + T - 1) / T, T, 0, stream>>>(W1, W1t, IN_CH, HIDHEADS);
  k_transpose_w<<<(HIDHEADS * OUT_CH + T - 1) / T, T, 0, stream>>>(W2, W2t, HIDHEADS, OUT_CH);

  // CSR build (once; reused by both layers)
  k_count  <<<(Etot + T - 1) / T, T, 0, stream>>>(dstE, E, N, deg);
  k_scan1  <<<nb, SCAN_B, 0, stream>>>(deg, N, incl, bsum);
  k_scan2  <<<1, 64, 0, stream>>>(bsum, boff, nb);
  k_scan3  <<<(N + T - 1) / T, T, 0, stream>>>(incl, deg, boff, N, starts, cursor);
  k_scatter<<<(Etot + T - 1) / T, T, 0, stream>>>(srcE, dstE, E, N, cursor, csr_src, epos);

  // ---- layer 1 ----
  {
    int tiles = (N / 16) * (HIDHEADS / 64);
    k_gemm_wmma<<<(tiles + 7) / 8, 256, 0, stream>>>(xh, W1t, b1, h1, N, IN_CH, HIDHEADS);
  }
  k_attn_scores<<<(N * NHEADS + T - 1) / T, T, 0, stream>>>(h1, as1, ad1, a_src1, a_dst1,
                                                            N, NHEADS, HIDHEADS / NHEADS);
  k_edge_exp<<<(Etot + T - 1) / T, T, 0, stream>>>(srcE, dstE, E, N, a_src1, a_dst1,
                                                   epos, csr_w1, NHEADS);
  _Float16* h2in = xh;  // xh dead after gemm1
  k_node_agg1<<<(N + 7) / 8, 256, 0, stream>>>(starts, deg, csr_src, csr_w1, h1, b1, h2in, N);

  // ---- layer 2 ----
  {
    int tiles = (N / 16) * (OUT_CH / 64);
    k_gemm_wmma<<<(tiles + 7) / 8, 256, 0, stream>>>(h2in, W2t, nullptr, h2, N, HIDHEADS, OUT_CH);
  }
  k_attn_scores<<<(N + T - 1) / T, T, 0, stream>>>(h2, as2, ad2, a_src2, a_dst2, N, 1, OUT_CH);
  k_edge_exp<<<(Etot + T - 1) / T, T, 0, stream>>>(srcE, dstE, E, N, a_src2, a_dst2,
                                                   epos, csr_w2, 1);
  k_node_agg2<<<(N + 7) / 8, 256, 0, stream>>>(starts, deg, csr_src, csr_w2, h2, b2, outf, N);
}
